// EncoderLayer_84164179133304
// MI455X (gfx1250) — matrix operand order repaired and checked by reading, not compiled, not run on hardware
//
#include <hip/hip_runtime.h>
#include <hip/hip_bf16.h>

typedef __attribute__((ext_vector_type(16))) _Float16 v16h;
typedef __attribute__((ext_vector_type(8)))  float    v8f;

#define B_    16
#define L_    1536
#define D_    512
#define DFF_  2048
#define MODES_ 64
#define NTOK  (B_ * L_)          // 24576
#define BM    128
#define BN    128
#define BK    32
#define LDT   40                 // padded LDS leading dim (halfs), dodges bank conflicts

union FragH { v16h v; _Float16 e[16]; };
union FragF { v8f  v; float    e[8];  };

// ---- WMMA fragment gathers from LDS, per CDNA5 ISA 7.12.2 16-bit layouts ----
// A (16x32, MxK): lanes 0-15 -> M=lane, VGPR v<4: K=2v,2v+1 ; v>=4: K=16+2(v-4)..
//                 lanes16-31 -> M=lane-16, K base +8.
__device__ inline v16h frag_a_from_lds(const _Float16* tile, int row0, int lane) {
  const _Float16* p = tile + (row0 + (lane & 15)) * LDT;
  const int kb = (lane < 16) ? 0 : 8;
  FragH f;
#pragma unroll
  for (int v = 0; v < 8; ++v) {
    const int k0 = (v < 4) ? (kb + 2 * v) : (16 + kb + 2 * (v - 4));
    f.e[2 * v]     = p[k0];
    f.e[2 * v + 1] = p[k0 + 1];
  }
  return f.v;
}
// B (32x16, KxN) stored in LDS as [n][k]: lanes 0-15 -> N=lane, VGPR v<4: K=2v,2v+1 ; v>=4: K=16+2(v-4)..
// lanes 16-31 -> N=lane-16, K base +8 (same k order as A).
__device__ inline v16h frag_b_from_lds(const _Float16* tile, int n0, int lane) {
  const _Float16* p = tile + (n0 + (lane & 15)) * LDT;
  const int kb = (lane < 16) ? 0 : 8;
  FragH f;
#pragma unroll
  for (int v = 0; v < 8; ++v) {
    const int k0 = (v < 4) ? (kb + 2 * v) : (16 + kb + 2 * (v - 4));
    f.e[2 * v]     = p[k0];
    f.e[2 * v + 1] = p[k0 + 1];
  }
  return f.v;
}

// ---- DFT bases: fwd (128 x 1536): rows 0-63 cos, rows 64-127 -sin.
//      inv (1536 x 128): cols scaled by 1/L (m=0) or 2/L.
__global__ void build_bases(float* __restrict__ fwdB, float* __restrict__ invB) {
  int idx = blockIdx.x * blockDim.x + threadIdx.x;
  if (idx >= 128 * L_) return;
  int mrow = idx / L_;
  int l    = idx - mrow * L_;
  int m    = mrow & 63;
  float ang = 6.283185307179586f * (float)((m * l) % L_) / (float)L_;
  float c = __cosf(ang), s = __sinf(ang);
  fwdB[idx] = (mrow < 64) ? c : -s;
  float sc = ((m == 0) ? 1.0f : 2.0f) / (float)L_;
  invB[(long long)l * 128 + mrow] = (mrow < 64) ? (sc * c) : (-sc * s);
}

// ---- Generic batched WMMA GEMM: C = act(A(MxK) * B + bias + R) ----
// B is (N,K) row-major if bIsKN==0, else (K,N) row-major.
// 256 threads = 8 wave32; wave (wm,wn) owns a 64x32 sub-tile (4x2 WMMA accs).
__global__ __launch_bounds__(256)
void gemm_wmma(const float* __restrict__ A, const float* __restrict__ Bm,
               const float* __restrict__ bias, const float* __restrict__ R,
               float* __restrict__ C,
               int M, int N, int K, int ldA, int ldB, int ldC,
               int bIsKN, int doRelu,
               long long sA, long long sB, long long sC, long long sR) {
  __shared__ _Float16 As[BM * LDT];
  __shared__ _Float16 Bs[BN * LDT];
  const int bz = blockIdx.z;
  A += (long long)bz * sA;
  Bm += (long long)bz * sB;
  C += (long long)bz * sC;
  if (R) R += (long long)bz * sR;

  const int tid  = threadIdx.x;
  const int lane = tid & 31;
  const int wave = tid >> 5;
  const int wm   = wave >> 2;          // 0..1
  const int wn   = wave & 3;           // 0..3
  const int m0   = blockIdx.y * BM;
  const int n0   = blockIdx.x * BN;

  const v8f vzero = {0.f, 0.f, 0.f, 0.f, 0.f, 0.f, 0.f, 0.f};
  v8f acc[4][2];
#pragma unroll
  for (int i = 0; i < 4; ++i)
#pragma unroll
    for (int j = 0; j < 2; ++j) acc[i][j] = vzero;

  for (int k0 = 0; k0 < K; k0 += BK) {
    // A tile: 128 x 32 f32 -> f16 LDS
#pragma unroll
    for (int i = 0; i < 4; ++i) {
      int idx = tid + i * 256;               // 0..1023 float4 slots
      int row = idx >> 3;
      int c4  = (idx & 7) << 2;
      const float4 a4 = *(const float4*)(A + (long long)(m0 + row) * ldA + k0 + c4);
      _Float16* d = As + row * LDT + c4;
      d[0] = (_Float16)a4.x; d[1] = (_Float16)a4.y;
      d[2] = (_Float16)a4.z; d[3] = (_Float16)a4.w;
    }
    // B tile -> LDS as [n][k]
    if (!bIsKN) {
#pragma unroll
      for (int i = 0; i < 4; ++i) {
        int idx = tid + i * 256;
        int row = idx >> 3;
        int c4  = (idx & 7) << 2;
        const float4 b4 = *(const float4*)(Bm + (long long)(n0 + row) * ldB + k0 + c4);
        _Float16* d = Bs + row * LDT + c4;
        d[0] = (_Float16)b4.x; d[1] = (_Float16)b4.y;
        d[2] = (_Float16)b4.z; d[3] = (_Float16)b4.w;
      }
    } else {
#pragma unroll
      for (int i = 0; i < 4; ++i) {
        int idx = tid + i * 256;
        int kr  = idx >> 5;                   // 0..31
        int n4  = (idx & 31) << 2;            // 0..124
        const float4 b4 = *(const float4*)(Bm + (long long)(k0 + kr) * ldB + n0 + n4);
        Bs[(n4 + 0) * LDT + kr] = (_Float16)b4.x;
        Bs[(n4 + 1) * LDT + kr] = (_Float16)b4.y;
        Bs[(n4 + 2) * LDT + kr] = (_Float16)b4.z;
        Bs[(n4 + 3) * LDT + kr] = (_Float16)b4.w;
      }
    }
    __syncthreads();

    v16h bf0 = frag_b_from_lds(Bs, wn * 32, lane);
    v16h bf1 = frag_b_from_lds(Bs, wn * 32 + 16, lane);
#pragma unroll
    for (int i = 0; i < 4; ++i) {
      v16h af = frag_a_from_lds(As, wm * 64 + i * 16, lane);
      acc[i][0] = __builtin_amdgcn_wmma_f32_16x16x32_f16(false, af, false, bf0,
                                                         (short)0, acc[i][0], false, false);
      acc[i][1] = __builtin_amdgcn_wmma_f32_16x16x32_f16(false, af, false, bf1,
                                                         (short)0, acc[i][1], false, false);
    }
    __syncthreads();
  }

  // Epilogue: C layout per ISA 7.12.2 (lanes 0-15: M=r, lanes16-31: M=8+r; N=lane&15)
#pragma unroll
  for (int i = 0; i < 4; ++i) {
#pragma unroll
    for (int j = 0; j < 2; ++j) {
      const int n = n0 + wn * 32 + j * 16 + (lane & 15);
      const float bv = bias ? bias[n] : 0.f;
      FragF fa; fa.v = acc[i][j];
#pragma unroll
      for (int r = 0; r < 8; ++r) {
        const int m = m0 + wm * 64 + i * 16 + r + ((lane < 16) ? 0 : 8);
        float v = fa.e[r] + bv;
        if (R) v += R[(long long)m * ldC + n];
        if (doRelu) v = fmaxf(v, 0.f);
        C[(long long)m * ldC + n] = v;
      }
    }
  }
}

// ---- Per-mode complex GEMM: yft[b,m,e] = sum_d xft[b,m,d] * (wr+i*wi)[d,e,m] ----
// M=16 (batch rows), N=128 per block, K=512.
// f16 WMMA has no A/B negation (NEG = {CNeg,0,0}), so the real part uses a
// separate accumulator for ai*bi and subtracts in the epilogue.
// wr/wi scaled by 4096 into f16 (they are O(1/D^2)); rescaled at store.
__global__ __launch_bounds__(256)
void mode_cgemm(const float* __restrict__ xft, const float* __restrict__ wr,
                const float* __restrict__ wi, float* __restrict__ yft) {
  __shared__ _Float16 Asr[16 * LDT], Asi[16 * LDT];
  __shared__ _Float16 Bsr[BN * LDT], Bsi[BN * LDT];
  const int m    = blockIdx.y;
  const int e0   = blockIdx.x * 128;
  const int tid  = threadIdx.x;
  const int lane = tid & 31;
  const int wave = tid >> 5;
  const float WSCL = 4096.0f;

  const v8f vzero = {0.f, 0.f, 0.f, 0.f, 0.f, 0.f, 0.f, 0.f};
  v8f accR = vzero, accN = vzero, accI = vzero;

  for (int k0 = 0; k0 < D_; k0 += 32) {
    {
      int row = tid >> 4;                     // b = 0..15
      int c2  = (tid & 15) << 1;
      const float2 vr = *(const float2*)(xft + ((long long)row * 128 + m) * D_ + k0 + c2);
      const float2 vi = *(const float2*)(xft + ((long long)row * 128 + 64 + m) * D_ + k0 + c2);
      Asr[row * LDT + c2]     = (_Float16)vr.x;
      Asr[row * LDT + c2 + 1] = (_Float16)vr.y;
      Asi[row * LDT + c2]     = (_Float16)vi.x;
      Asi[row * LDT + c2 + 1] = (_Float16)vi.y;
    }
#pragma unroll
    for (int i = 0; i < 16; ++i) {
      int idx = tid + i * 256;                // 0..4095
      int e   = idx & 127;
      int d   = idx >> 7;                     // 0..31
      long long g = ((long long)(k0 + d) * D_ + (e0 + e)) * MODES_ + m;
      Bsr[e * LDT + d] = (_Float16)(wr[g] * WSCL);
      Bsi[e * LDT + d] = (_Float16)(wi[g] * WSCL);
    }
    __syncthreads();

    v16h ar = frag_a_from_lds(Asr, 0, lane);
    v16h ai = frag_a_from_lds(Asi, 0, lane);
    v16h br = frag_b_from_lds(Bsr, wave * 16, lane);
    v16h bi = frag_b_from_lds(Bsi, wave * 16, lane);
    accR = __builtin_amdgcn_wmma_f32_16x16x32_f16(false, ar, false, br, (short)0, accR, false, false);
    accN = __builtin_amdgcn_wmma_f32_16x16x32_f16(false, ai, false, bi, (short)0, accN, false, false);
    accI = __builtin_amdgcn_wmma_f32_16x16x32_f16(false, ar, false, bi, (short)0, accI, false, false);
    accI = __builtin_amdgcn_wmma_f32_16x16x32_f16(false, ai, false, br, (short)0, accI, false, false);
    __syncthreads();
  }

  const float unscale = 1.0f / 4096.0f;
  const int e = e0 + wave * 16 + (lane & 15);
  FragF fr, fn, fi; fr.v = accR; fn.v = accN; fi.v = accI;
#pragma unroll
  for (int r = 0; r < 8; ++r) {
    const int b = r + ((lane < 16) ? 0 : 8);
    yft[((long long)b * 128 + m) * D_ + e]      = (fr.e[r] - fn.e[r]) * unscale;
    yft[((long long)b * 128 + 64 + m) * D_ + e] = fi.e[r] * unscale;
  }
}

// ---- 256 -> 3 projection per token ----
__global__ void logits3_kernel(const float* __restrict__ h, const float* __restrict__ w2,
                               const float* __restrict__ b2, float* __restrict__ lg) {
  int t = blockIdx.x * blockDim.x + threadIdx.x;
  if (t >= NTOK) return;
  const float* hp = h + (long long)t * 256;
  float s0 = 0.f, s1 = 0.f, s2 = 0.f;
#pragma unroll 4
  for (int i = 0; i < 256; ++i) {
    float hv = hp[i];
    s0 += hv * w2[i];
    s1 += hv * w2[256 + i];
    s2 += hv * w2[512 + i];
  }
  float4 o;
  o.x = s0 + b2[0]; o.y = s1 + b2[1]; o.z = s2 + b2[2]; o.w = 0.f;
  *(float4*)(lg + 4 * (long long)t) = o;
}

// ---- Fused softmax gate + (3,5,7) avg pools + seasonal subtract ----
// 7-tap window with per-offset weight g0/3 (|o|<=1) + g1/5 (|o|<=2) + g2/7.
__global__ void decomp_combine(const float* __restrict__ z, const float* __restrict__ lg,
                               float* __restrict__ out) {
  const int DV = D_ / 4;                          // 128 float4 per token
  long long idx = (long long)blockIdx.x * blockDim.x + threadIdx.x;
  if (idx >= (long long)NTOK * DV) return;
  int t  = (int)(idx / DV);
  int d4 = (int)(idx - (long long)t * DV) * 4;
  int b  = t / L_;
  int l  = t - b * L_;

  float g0 = lg[4 * (long long)t], g1 = lg[4 * (long long)t + 1], g2 = lg[4 * (long long)t + 2];
  float mx = fmaxf(g0, fmaxf(g1, g2));
  g0 = __expf(g0 - mx); g1 = __expf(g1 - mx); g2 = __expf(g2 - mx);
  float inv = 1.0f / (g0 + g1 + g2);
  g0 *= inv; g1 *= inv; g2 *= inv;

  const float* base = z + ((long long)b * L_) * D_ + d4;
  float a0 = 0.f, a1 = 0.f, a2 = 0.f, a3 = 0.f;
#pragma unroll
  for (int o = -3; o <= 3; ++o) {
    int ll = l + o;
    if (ll < 0 || ll >= L_) continue;
    int ao = (o < 0) ? -o : o;
    float w = g2 * (1.0f / 7.0f);
    if (ao <= 2) w += g1 * (1.0f / 5.0f);
    if (ao <= 1) w += g0 * (1.0f / 3.0f);
    const float* p = base + (long long)ll * D_;
    a0 += w * p[0]; a1 += w * p[1]; a2 += w * p[2]; a3 += w * p[3];
  }
  const float* pc = base + (long long)l * D_;
  float4 o4;
  o4.x = pc[0] - a0; o4.y = pc[1] - a1; o4.z = pc[2] - a2; o4.w = pc[3] - a3;
  *(float4*)(out + ((long long)b * L_ + l) * D_ + d4) = o4;
}

extern "C" void kernel_launch(void* const* d_in, const int* in_sizes, int n_in,
                              void* d_out, int out_size, void* d_ws, size_t ws_size,
                              hipStream_t stream) {
  (void)in_sizes; (void)n_in; (void)out_size; (void)ws_size;
  const float* x      = (const float*)d_in[0];
  const float* w_real = (const float*)d_in[1];
  const float* w_imag = (const float*)d_in[2];
  const float* conv1  = (const float*)d_in[3];   // (2048, 512)  = (N,K)
  const float* conv2  = (const float*)d_in[4];   // (512, 2048)  = (N,K)
  const float* d1w1   = (const float*)d_in[5];   // (256, 512)   = (N,K)
  const float* d1b1   = (const float*)d_in[6];
  const float* d1w2   = (const float*)d_in[7];   // (3, 256)
  const float* d1b2   = (const float*)d_in[8];
  const float* d2w1   = (const float*)d_in[9];
  const float* d2b1   = (const float*)d_in[10];
  const float* d2w2   = (const float*)d_in[11];
  const float* d2b2   = (const float*)d_in[12];
  float* out = (float*)d_out;

  float* ws = (float*)d_ws;
  size_t off = 0;
  auto alloc = [&](size_t n) { float* p = ws + off; off += n; return p; };
  float* fwdB = alloc((size_t)128 * L_);          // forward DFT basis (128 x 1536)
  float* invB = alloc((size_t)L_ * 128);          // inverse DFT basis (1536 x 128)
  float* xft  = alloc((size_t)B_ * 128 * D_);     // [b][re0..63,im0..63][d]
  float* yft  = alloc((size_t)B_ * 128 * D_);
  float* x1   = alloc((size_t)NTOK * D_);         // x + fourier(x)
  float* s1   = alloc((size_t)NTOK * D_);         // seasonal 1
  float* h    = alloc((size_t)NTOK * 256);        // gating hidden (reused)
  float* lg   = alloc((size_t)NTOK * 4);          // logits, stride 4
  float* y1   = alloc((size_t)NTOK * DFF_);       // FFN hidden
  float* z2   = alloc((size_t)NTOK * D_);         // s1 + FFN(s1)

  const long long sBL = (long long)L_ * D_;       // per-batch stride of (L,D)
  const long long sFT = (long long)128 * D_;      // per-batch stride of (128,D)

  // 1) DFT bases
  build_bases<<<(128 * L_ + 255) / 256, 256, 0, stream>>>(fwdB, invB);

  // 2) forward truncated DFT: xft[b] = fwdB(128x1536) @ x[b](1536x512, K-major)
  gemm_wmma<<<dim3(D_ / BN, 128 / BM, B_), 256, 0, stream>>>(
      fwdB, x, nullptr, nullptr, xft, 128, D_, L_, L_, D_, D_,
      /*bIsKN=*/1, /*relu=*/0, 0, sBL, sFT, 0);

  // 3) per-mode complex GEMM
  mode_cgemm<<<dim3(D_ / BN, MODES_), 256, 0, stream>>>(xft, w_real, w_imag, yft);

  // 4) inverse DFT + residual: x1[b] = invB(1536x128) @ yft[b](128x512) + x[b]
  gemm_wmma<<<dim3(D_ / BN, L_ / BM, B_), 256, 0, stream>>>(
      invB, yft, nullptr, x, x1, L_, D_, 128, 128, D_, D_,
      /*bIsKN=*/1, /*relu=*/0, 0, sFT, sBL, sBL);

  // 5) decomp 1: gating MLP + fused pool/softmax/subtract
  gemm_wmma<<<dim3(256 / BN, NTOK / BM, 1), 256, 0, stream>>>(
      x1, d1w1, d1b1, nullptr, h, NTOK, 256, D_, D_, D_, 256,
      /*bIsKN=*/0, /*relu=*/1, 0, 0, 0, 0);
  logits3_kernel<<<(NTOK + 255) / 256, 256, 0, stream>>>(h, d1w2, d1b2, lg);
  decomp_combine<<<(int)(((long long)NTOK * (D_ / 4) + 255) / 256), 256, 0, stream>>>(x1, lg, s1);

  // 6) FFN: y1 = relu(s1 @ conv1^T); z2 = y1 @ conv2^T + s1
  gemm_wmma<<<dim3(DFF_ / BN, NTOK / BM, 1), 256, 0, stream>>>(
      s1, conv1, nullptr, nullptr, y1, NTOK, DFF_, D_, D_, D_, DFF_,
      /*bIsKN=*/0, /*relu=*/1, 0, 0, 0, 0);
  gemm_wmma<<<dim3(D_ / BN, NTOK / BM, 1), 256, 0, stream>>>(
      y1, conv2, nullptr, s1, z2, NTOK, D_, DFF_, DFF_, DFF_, D_,
      /*bIsKN=*/0, /*relu=*/0, 0, 0, 0, 0);

  // 7) decomp 2 -> output
  gemm_wmma<<<dim3(256 / BN, NTOK / BM, 1), 256, 0, stream>>>(
      z2, d2w1, d2b1, nullptr, h, NTOK, 256, D_, D_, D_, 256,
      /*bIsKN=*/0, /*relu=*/1, 0, 0, 0, 0);
  logits3_kernel<<<(NTOK + 255) / 256, 256, 0, stream>>>(h, d2w2, d2b2, lg);
  decomp_combine<<<(int)(((long long)NTOK * (D_ / 4) + 255) / 256), 256, 0, stream>>>(z2, lg, out);
}
